// VQVAE_84404697301209
// MI455X (gfx1250) — matrix operand order repaired and checked by reading, not compiled, not run on hardware
//
#include <hip/hip_runtime.h>
#include <hip/hip_bf16.h>

// VQ-VAE nearest-codebook quantization, fused for MI455X (gfx1250, wave32).
// argmin_k ||z - c_k|| == argmin_k (||c_k||^2 - 2 z.c_k)  -> one f16 WMMA GEMM,
// fused argmin, fused gather + commitment loss. No [B,N,K] tensor materialized.

typedef __attribute__((ext_vector_type(16))) _Float16 v16h;
typedef __attribute__((ext_vector_type(4)))  _Float16 v4h;
typedef __attribute__((ext_vector_type(8)))  float    v8f;

#define BQ 16
#define NQ 4096
#define DQ 64
#define KQ 1024
#define KC 256                         // codebook rows staged in LDS per chunk
#define WAVES 8
#define ROWS_PER_WG (WAVES * 16)       // 128 z-rows per workgroup
#define NBLOCKS ((BQ * NQ) / ROWS_PER_WG)  // 512

__global__ __launch_bounds__(256)
void vq_main(const float* __restrict__ z, const float* __restrict__ cb,
             float* __restrict__ out, float* __restrict__ partials) {
    __shared__ __align__(32) _Float16 cb_lds[KC * DQ];  // 32 KB f16 codebook chunk
    __shared__ float csq_lds[KC];                       // ||c||^2 per staged row
    __shared__ int   idx_lds[WAVES * 16];
    __shared__ float loss_lds[WAVES];

    const int tid   = threadIdx.x;
    const int lane  = tid & 31;
    const int wave  = tid >> 5;
    const int nlane = lane & 15;   // column (B/C/D N index) within tile
    const int sel   = lane >> 4;   // half-wave select for A/B K-striping

    const int rowbase = blockIdx.x * ROWS_PER_WG + wave * 16;

    // ---- A fragments: one 16x64 z tile, f32->f16, 16x16x32 WMMA A layout ----
    // lanes 0-15: K 0..15; lanes 16-31: K 16..31 (then +32 for A1)
    const float* za = z + (size_t)(rowbase + nlane) * DQ;
    v16h A0, A1;
#pragma unroll
    for (int j = 0; j < 8; ++j) {
        A0[j]     = (_Float16)za[sel * 16 + j];
        A0[j + 8] = (_Float16)za[8 + sel * 16 + j];
        A1[j]     = (_Float16)za[32 + sel * 16 + j];
        A1[j + 8] = (_Float16)za[40 + sel * 16 + j];
    }

    float bestv[8];
    int   besti[8];
#pragma unroll
    for (int r = 0; r < 8; ++r) { bestv[r] = 3.4e38f; besti[r] = 0; }

    for (int c = 0; c < KQ / KC; ++c) {
        __syncthreads();  // previous chunk's readers done before restaging
        {   // ---- stage KC codebook rows into LDS (f16) + row norms ----
            const float* crow = cb + (size_t)(c * KC + tid) * DQ;
            float sq = 0.0f;
#pragma unroll
            for (int j = 0; j < DQ / 4; ++j) {
                float4 v = ((const float4*)crow)[j];
                sq += v.x * v.x + v.y * v.y + v.z * v.z + v.w * v.w;
                v4h h; h[0] = (_Float16)v.x; h[1] = (_Float16)v.y;
                       h[2] = (_Float16)v.z; h[3] = (_Float16)v.w;
                *(v4h*)&cb_lds[tid * DQ + j * 4] = h;
            }
            csq_lds[tid] = sq;
        }
        __syncthreads();

        // ---- 16 column-tiles of 16 codebook entries; K=64 via two WMMAs ----
        for (int kt = 0; kt < KC / 16; ++kt) {
            const int col_local = kt * 16 + nlane;
            const _Float16* bp = &cb_lds[col_local * DQ];
            // B 32x16 layout: lanes 0-15 hold K=0..15, lanes 16-31 hold K=16..31
            v16h B0 = *(const v16h*)(bp + sel * 16);        // K 0..31
            v16h B1 = *(const v16h*)(bp + 32 + sel * 16);   // K 32..63
            v8f acc = {};
            acc = __builtin_amdgcn_wmma_f32_16x16x32_f16(false, A0, false, B0,
                                                         (short)0, acc, false, false);
            acc = __builtin_amdgcn_wmma_f32_16x16x32_f16(false, A1, false, B1,
                                                         (short)0, acc, false, false);
            const float csq  = csq_lds[col_local];
            const int   gcol = c * KC + kt * 16 + nlane;
#pragma unroll
            for (int r = 0; r < 8; ++r) {
                float score = csq - 2.0f * acc[r];  // argmin-equivalent to distance
                if (score < bestv[r]) { bestv[r] = score; besti[r] = gcol; }
            }
        }
    }

    // ---- cross-lane argmin: 16 lanes hold the 16 columns of each row ----
#pragma unroll
    for (int r = 0; r < 8; ++r) {
        float v = bestv[r];
        int   i = besti[r];
#pragma unroll
        for (int m = 8; m >= 1; m >>= 1) {
            float ov = __shfl_xor(v, m, 32);
            int   oi = __shfl_xor(i, m, 32);
            if (ov < v || (ov == v && oi < i)) { v = ov; i = oi; }
        }
        // lanes 0-15 carry row r, lanes 16-31 carry row r+8
        if (nlane == 0) idx_lds[wave * 16 + r + 8 * sel] = i;
    }
    __syncthreads();

    // ---- gather winning codebook rows, write z_q_st, commitment loss ----
    const int orow = lane >> 1;           // 2 lanes per output row
    const int part = lane & 1;            // each handles 32 of 64 floats
    const int bidx = idx_lds[wave * 16 + orow];
    const size_t g = (size_t)(rowbase + orow) * DQ + part * 32;
    const float* crow = cb + (size_t)bidx * DQ + part * 32;
    const float* zr   = z + g;
    float*       op   = out + g;
    float lsum = 0.0f;
#pragma unroll
    for (int j = 0; j < 8; ++j) {
        float4 cv = ((const float4*)crow)[j];
        float4 zv = ((const float4*)zr)[j];
        float dx = cv.x - zv.x, dy = cv.y - zv.y;
        float dz = cv.z - zv.z, dw = cv.w - zv.w;
        lsum += dx * dx + dy * dy + dz * dz + dw * dw;
        ((float4*)op)[j] = cv;
    }
#pragma unroll
    for (int m = 16; m >= 1; m >>= 1) lsum += __shfl_xor(lsum, m, 32);
    if (lane == 0) loss_lds[wave] = lsum;
    __syncthreads();
    if (tid == 0) {
        float s = 0.0f;
#pragma unroll
        for (int w = 0; w < WAVES; ++w) s += loss_lds[w];
        partials[blockIdx.x] = s;
    }
}

// Deterministic fixed-order reduction of the 512 per-block partials.
__global__ __launch_bounds__(256)
void vq_reduce(const float* __restrict__ partials, float* __restrict__ loss_out) {
    __shared__ float s[256];
    const int t = threadIdx.x;
    s[t] = partials[t] + partials[t + 256];
    __syncthreads();
    for (int w = 128; w > 0; w >>= 1) {
        if (t < w) s[t] += s[t + w];
        __syncthreads();
    }
    if (t == 0) *loss_out = s[0] * (1.0f / (float)(BQ * NQ * DQ));
}

extern "C" void kernel_launch(void* const* d_in, const int* in_sizes, int n_in,
                              void* d_out, int out_size, void* d_ws, size_t ws_size,
                              hipStream_t stream) {
    const float* z  = (const float*)d_in[0];   // [16,4096,64] f32
    const float* cb = (const float*)d_in[1];   // [1024,64]    f32
    float* out      = (float*)d_out;           // z_q_st flat + loss scalar
    float* partials = (float*)d_ws;            // 512 floats of scratch

    vq_main<<<NBLOCKS, 256, 0, stream>>>(z, cb, out, partials);
    vq_reduce<<<1, 256, 0, stream>>>(partials, out + (size_t)BQ * NQ * DQ);
}
